// SlotAttention_77970836291996
// MI455X (gfx1250) — matrix operand order repaired and checked by reading, not compiled, not run on hardware
//
#include <hip/hip_runtime.h>
#include <hip/hip_bf16.h>
#include <math.h>

typedef _Float16 half_t;
typedef __attribute__((ext_vector_type(16))) _Float16 v16h;
typedef __attribute__((ext_vector_type(8)))  _Float16 v8h;
typedef __attribute__((ext_vector_type(8)))  float    v8f;

#define Bn 4
#define Hn 64
#define Wn 64
#define Cn 256
#define Kn 5
#define Dsn 256
#define Nn 4096           // H*W
#define BN 16384          // B*N
#define BKN 81920         // B*K*N
#define LN_EPS 1e-5f

// ---------------- reductions ----------------
__device__ __forceinline__ float wave_sum(float v) {
  for (int o = 16; o > 0; o >>= 1) v += __shfl_xor(v, o, 32);
  return v;
}
__device__ __forceinline__ float wave_max(float v) {
  for (int o = 16; o > 0; o >>= 1) v = fmaxf(v, __shfl_xor(v, o, 32));
  return v;
}
__device__ __forceinline__ float wave_min(float v) {
  for (int o = 16; o > 0; o >>= 1) v = fminf(v, __shfl_xor(v, o, 32));
  return v;
}
__device__ __forceinline__ float block_sum256(float v, float* red) {
  v = wave_sum(v);
  if ((threadIdx.x & 31) == 0) red[threadIdx.x >> 5] = v;
  __syncthreads();
  float t = 0.f;
  #pragma unroll
  for (int i = 0; i < 8; ++i) t += red[i];
  __syncthreads();
  return t;
}
__device__ __forceinline__ float block_max256(float v, float* red) {
  v = wave_max(v);
  if ((threadIdx.x & 31) == 0) red[threadIdx.x >> 5] = v;
  __syncthreads();
  float t = -3.4e38f;
  #pragma unroll
  for (int i = 0; i < 8; ++i) t = fmaxf(t, red[i]);
  __syncthreads();
  return t;
}
__device__ __forceinline__ float block_min256(float v, float* red) {
  v = wave_min(v);
  if ((threadIdx.x & 31) == 0) red[threadIdx.x >> 5] = v;
  __syncthreads();
  float t = 3.4e38f;
  #pragma unroll
  for (int i = 0; i < 8; ++i) t = fminf(t, red[i]);
  __syncthreads();
  return t;
}

// ---------------- WMMA fragment helpers (ISA 7.12.2 layouts) ----------------
// A 16x32 f16: lane holds row M = lane&15; halves i<8 -> K = kBase+hi8+i,
// i>=8 -> K = kBase+16+hi8+(i-8), hi8 = (lane&16)?8:0.
__device__ __forceinline__ v16h load_a_frag(const half_t* A, int ldk, int mBase,
                                            int kBase, int lane) {
  const int m   = mBase + (lane & 15);
  const int hi8 = (lane & 16) ? 8 : 0;
  const half_t* p = A + (size_t)m * ldk + kBase + hi8;
  union { v16h v; v8h h[2]; } u;
  u.h[0] = *(const v8h*)(p);
  u.h[1] = *(const v8h*)(p + 16);
  return u.v;
}
// B 32x16 f16 where B[k][n] = W[n][k] (W row-major [n][k]): lane holds col
// N = nBase + (lane&15); i<8 -> K = kBase+hi8+i, i>=8 -> K = kBase+16+hi8+(i-8), hi8=(lane&16)?8:0.
__device__ __forceinline__ v16h load_b_frag(const half_t* W, int ldk, int nBase,
                                            int kBase, int lane) {
  const int n    = nBase + (lane & 15);
  const int hi8 = (lane & 16) ? 8 : 0;
  const half_t* p = W + (size_t)n * ldk + kBase + hi8;
  union { v16h v; v8h h[2]; } u;
  u.h[0] = *(const v8h*)(p);
  u.h[1] = *(const v8h*)(p + 16);
  return u.v;
}
// C/D f32 16x16: VGPR r -> M = mBase + r + ((lane&16)?8:0), N = nBase + (lane&15)
__device__ __forceinline__ void store_c_f16(half_t* D, int ld, int mBase, int nBase,
                                            int lane, v8f acc) {
  const int n    = nBase + (lane & 15);
  const int mOff = (lane & 16) ? 8 : 0;
  #pragma unroll
  for (int r = 0; r < 8; ++r)
    D[(size_t)(mBase + mOff + r) * ld + n] = (half_t)acc[r];
}

// ---------------- small prep kernels ----------------
__global__ void cvt_f32_to_f16(const float* __restrict__ src,
                               half_t* __restrict__ dst, int n) {
  int i = blockIdx.x * blockDim.x + threadIdx.x;
  if (i < n) dst[i] = (half_t)src[i];
}

// LayerNorm(feat) with norm_g/b -> f16, one block per row of 256
__global__ void ln_feat_kernel(const float* __restrict__ feat,
                               const float* __restrict__ g,
                               const float* __restrict__ b,
                               half_t* __restrict__ out) {
  __shared__ float red[8];
  const size_t row = blockIdx.x;
  const int c = threadIdx.x;
  float v = feat[row * Cn + c];
  float s  = block_sum256(v, red);
  float sq = block_sum256(v * v, red);
  float mean = s * (1.0f / Cn);
  float var  = sq * (1.0f / Cn) - mean * mean;
  float inv  = rsqrtf(var + LN_EPS);
  out[row * Cn + c] = (half_t)((v - mean) * inv * g[c] + b[c]);
}

// fg_pos: mask-weighted grid mean per k; also broadcast into d_out fg section
__global__ void fgpos_kernel(const float* __restrict__ mask,
                             float* __restrict__ fgpos,
                             float* __restrict__ out_fg) {
  __shared__ float red[8];
  const int k = blockIdx.x;
  float sx = 0.f, sy = 0.f, sm = 0.f;
  for (int n = threadIdx.x; n < Nn; n += 256) {
    float m = mask[k * Nn + n];
    int h = n >> 6, w = n & 63;
    float gx = -1.0f + (2.0f * w + 1.0f) / (float)Wn;
    float gy = -1.0f + (2.0f * h + 1.0f) / (float)Hn;
    sx += gx * m; sy += gy * m; sm += m;
  }
  sx = block_sum256(sx, red);
  sy = block_sum256(sy, red);
  sm = block_sum256(sm, red);
  if (threadIdx.x == 0) {
    float fx = sx / (sm + 1e-5f), fy = sy / (sm + 1e-5f);
    fgpos[k * 2 + 0] = fx; fgpos[k * 2 + 1] = fy;
    for (int b = 0; b < Bn; ++b) {
      out_fg[(b * Kn + k) * 2 + 0] = fx;
      out_fg[(b * Kn + k) * 2 + 1] = fy;
    }
  }
}

__global__ void slot_init_kernel(const float* __restrict__ mu,
                                 const float* __restrict__ logsigma,
                                 const float* __restrict__ noise,
                                 float* __restrict__ slot) {
  int i = blockIdx.x * blockDim.x + threadIdx.x;
  if (i < Bn * Kn * Dsn) {
    int d = i & (Dsn - 1);
    slot[i] = mu[d] + __expf(logsigma[d]) * noise[i];
  }
}

// ---------------- WMMA GEMM: D[m,n] = sum_k A[m,k] * W[n,k]  (all f16) ------
// grid: (M/32, Ndim/64), block 256 = 8 waves; each wave one 16x16 tile
__global__ void gemm_wt_f16(const half_t* __restrict__ A,
                            const half_t* __restrict__ W,
                            half_t* __restrict__ D, int Kdim, int Ndim) {
  const int wave = threadIdx.x >> 5;
  const int lane = threadIdx.x & 31;
  const int mBase = blockIdx.x * 32 + (wave >> 2) * 16;
  const int nBase = blockIdx.y * 64 + (wave & 3) * 16;
  v8f acc = {};
  for (int kc = 0; kc < Kdim; kc += 32) {
    v16h a = load_a_frag(A, Kdim, mBase, kc, lane);
    v16h b = load_b_frag(W, Kdim, nBase, kc, lane);
    acc = __builtin_amdgcn_wmma_f32_16x16x32_f16(false, a, false, b,
                                                 (short)0, acc, false, false);
  }
  store_c_f16(D, Ndim, mBase, nBase, lane, acc);
}

// ------- fused: t = base + ge(relpos); LN(mlp_g/b); @ mlp_W.T + bias --------
// one block per 16 consecutive rows of (b,k,n); LDS-staged A tile + WMMA
__global__ void kkvv_kernel(const half_t* __restrict__ base,   // (B*N, C) f16
                            const half_t* __restrict__ mlpW_h, // (Ds, C) f16
                            const float* __restrict__ grid_W,  // (C,4)
                            const float* __restrict__ grid_b,
                            const float* __restrict__ mlp_g,
                            const float* __restrict__ mlp_b,
                            const float* __restrict__ mlp_bias,
                            const float* __restrict__ fgpos,   // (K,2)
                            half_t* __restrict__ out) {        // (B*K*N, Ds)
  __shared__ half_t tk[16 * Cn];   // 8 KB LN'd activation tile
  const int r0 = blockIdx.x * 16;
  const int bk = r0 >> 12;              // /N
  const int b  = bk / Kn;
  const int k  = bk % Kn;
  const int nBase = r0 & (Nn - 1);

  // phase 1: 16 threads per row, 16 cols each; LN with 16-lane shuffles
  {
    const int r = threadIdx.x >> 4;     // row in tile
    const int s = threadIdx.x & 15;     // col group
    const int n = nBase + r;
    const int h = n >> 6, w = n & 63;
    const float gx = -1.0f + (2.0f * w + 1.0f) / (float)Wn;
    const float gy = -1.0f + (2.0f * h + 1.0f) / (float)Hn;
    const float relx = gx - fgpos[k * 2 + 0];
    const float rely = gy - fgpos[k * 2 + 1];
    const half_t* brow = base + ((size_t)b * Nn + n) * Cn;
    float tv[16];
    float sum = 0.f, sq = 0.f;
    #pragma unroll
    for (int j = 0; j < 16; ++j) {
      const int c = s * 16 + j;
      const float4 gw = ((const float4*)grid_W)[c];
      float t = (float)brow[c] + relx * (gw.x - gw.z) + rely * (gw.y - gw.w)
                + grid_b[c];
      tv[j] = t; sum += t; sq += t * t;
    }
    #pragma unroll
    for (int o = 1; o < 16; o <<= 1) {
      sum += __shfl_xor(sum, o, 16);
      sq  += __shfl_xor(sq,  o, 16);
    }
    const float mean = sum * (1.0f / Cn);
    const float var  = sq * (1.0f / Cn) - mean * mean;
    const float inv  = rsqrtf(var + LN_EPS);
    #pragma unroll
    for (int j = 0; j < 16; ++j) {
      const int c = s * 16 + j;
      tk[r * Cn + c] = (half_t)((tv[j] - mean) * inv * mlp_g[c] + mlp_b[c]);
    }
  }
  __syncthreads();

  // phase 2: 8 waves x 2 col-tiles of 16; A from LDS, B from global mlp_W
  const int wave = threadIdx.x >> 5;
  const int lane = threadIdx.x & 31;
  half_t* outBase = out + ((size_t)bk * Nn + nBase) * Dsn;
  for (int tt = 0; tt < 2; ++tt) {
    const int dBase = (wave * 2 + tt) * 16;
    v8f acc = {};
    #pragma unroll
    for (int kc = 0; kc < Cn; kc += 32) {
      v16h a = load_a_frag(tk, Cn, 0, kc, lane);
      v16h bb = load_b_frag(mlpW_h, Cn, dBase, kc, lane);
      acc = __builtin_amdgcn_wmma_f32_16x16x32_f16(false, a, false, bb,
                                                   (short)0, acc, false, false);
    }
    const float bias = mlp_bias[dBase + (lane & 15)];
    #pragma unroll
    for (int r = 0; r < 8; ++r) acc[r] += bias;
    store_c_f16(outBase, Dsn, 0, dBase, lane, acc);
  }
}

// ---------------- iteration kernels ----------------
// q = LN(slot; q_g,q_b) @ q_W.T ; one block per (b,k)
__global__ void q_kernel(const float* __restrict__ slot,
                         const float* __restrict__ q_g,
                         const float* __restrict__ q_b,
                         const float* __restrict__ q_W,
                         float* __restrict__ q) {
  __shared__ float sq_[Dsn];
  __shared__ float red[8];
  const int bk = blockIdx.x;
  const int d = threadIdx.x;
  float v = slot[bk * Dsn + d];
  float s  = block_sum256(v, red);
  float s2 = block_sum256(v * v, red);
  float mean = s * (1.0f / Dsn);
  float var  = s2 * (1.0f / Dsn) - mean * mean;
  sq_[d] = (v - mean) * rsqrtf(var + LN_EPS) * q_g[d] + q_b[d];
  __syncthreads();
  float acc = 0.f;
  const float* wrow = q_W + (size_t)d * Dsn;
  for (int c = 0; c < Dsn; ++c) acc += sq_[c] * wrow[c];
  q[bk * Dsn + d] = acc;
}

// attn + softmax + upd, one block per (b,k)
__global__ void attn_kernel(const float* __restrict__ q,
                            const half_t* __restrict__ kk,
                            const half_t* __restrict__ vv,
                            const float* __restrict__ mask,
                            float* __restrict__ attn_buf,
                            float* __restrict__ upd) {
  __shared__ float at[Nn];     // 16 KB
  __shared__ float qs[Dsn];
  __shared__ float red[8];
  const int bk = blockIdx.x;
  const int k = bk % Kn;
  const float scale = 0.0625f;  // 256^-0.5
  qs[threadIdx.x] = q[bk * Dsn + threadIdx.x];
  __syncthreads();

  float mx = -3.4e38f;
  for (int i = 0; i < 16; ++i) {
    const int n = threadIdx.x + i * 256;
    const half_t* row = kk + ((size_t)bk * Nn + n) * Dsn;
    __builtin_prefetch(row + Dsn, 0, 3);
    float s = 0.f;
    #pragma unroll
    for (int c8 = 0; c8 < Dsn / 8; ++c8) {
      v8h hv = ((const v8h*)row)[c8];
      #pragma unroll
      for (int j = 0; j < 8; ++j) s += qs[c8 * 8 + j] * (float)hv[j];
    }
    s *= scale;
    if (mask[k * Nn + n] == 0.0f) s = -1e9f;
    at[n] = s;
    mx = fmaxf(mx, s);
  }
  mx = block_max256(mx, red);
  float lsum = 0.f;
  for (int i = 0; i < 16; ++i) {
    const int n = threadIdx.x + i * 256;
    float e = __expf(at[n] - mx);
    at[n] = e;
    lsum += e;
  }
  float tot = block_sum256(lsum, red);
  const float inv = 1.0f / tot;
  for (int i = 0; i < 16; ++i) {
    const int n = threadIdx.x + i * 256;
    float a = at[n] * inv;
    at[n] = a;
    attn_buf[(size_t)bk * Nn + n] = a;
  }
  __syncthreads();
  // upd[d] = sum_n at[n] * vv[n,d]
  const int d = threadIdx.x;
  float u = 0.f;
  const half_t* vb = vv + (size_t)bk * Nn * Dsn + d;
  for (int n = 0; n < Nn; ++n) u += at[n] * (float)vb[(size_t)n * Dsn];
  upd[bk * Dsn + d] = u;
}

// GRU cell + residual MLP, one block per (b,k)
__global__ void gru_kernel(const float* __restrict__ upd,
                           const float* __restrict__ slot_in,
                           const float* __restrict__ Wih,
                           const float* __restrict__ Whh,
                           const float* __restrict__ bih,
                           const float* __restrict__ bhh,
                           const float* __restrict__ res_g,
                           const float* __restrict__ res_b,
                           const float* __restrict__ res_W,
                           const float* __restrict__ res_bias,
                           float* __restrict__ slot_out) {
  __shared__ float xs[Dsn], hs[Dsn], lnh[Dsn];
  __shared__ float red[8];
  const int bk = blockIdx.x;
  const int d = threadIdx.x;
  xs[d] = upd[bk * Dsn + d];
  hs[d] = slot_in[bk * Dsn + d];
  __syncthreads();
  float gir = 0.f, giz = 0.f, gin = 0.f, ghr = 0.f, ghz = 0.f, ghn = 0.f;
  const float* wir = Wih + (size_t)d * Dsn;
  const float* wiz = Wih + (size_t)(Dsn + d) * Dsn;
  const float* win = Wih + (size_t)(2 * Dsn + d) * Dsn;
  const float* whr = Whh + (size_t)d * Dsn;
  const float* whz = Whh + (size_t)(Dsn + d) * Dsn;
  const float* whn = Whh + (size_t)(2 * Dsn + d) * Dsn;
  for (int c = 0; c < Dsn; ++c) {
    const float x = xs[c], h = hs[c];
    gir += x * wir[c]; giz += x * wiz[c]; gin += x * win[c];
    ghr += h * whr[c]; ghz += h * whz[c]; ghn += h * whn[c];
  }
  gir += bih[d];           giz += bih[Dsn + d];     gin += bih[2 * Dsn + d];
  ghr += bhh[d];           ghz += bhh[Dsn + d];     ghn += bhh[2 * Dsn + d];
  const float r = 1.0f / (1.0f + __expf(-(gir + ghr)));
  const float z = 1.0f / (1.0f + __expf(-(giz + ghz)));
  const float nn = tanhf(gin + r * ghn);
  const float hnew = (1.0f - z) * nn + z * hs[d];
  // LN(hnew) with res_g/b
  float s  = block_sum256(hnew, red);
  float s2 = block_sum256(hnew * hnew, red);
  float mean = s * (1.0f / Dsn);
  float var  = s2 * (1.0f / Dsn) - mean * mean;
  lnh[d] = (hnew - mean) * rsqrtf(var + LN_EPS) * res_g[d] + res_b[d];
  __syncthreads();
  float acc = 0.f;
  const float* wr = res_W + (size_t)d * Dsn;
  for (int c = 0; c < Dsn; ++c) acc += lnh[c] * wr[c];
  slot_out[bk * Dsn + d] = acc + res_bias[d] + hs[d];
}

// ---------------- output kernels ----------------
__global__ void copy_slot_kernel(const float* __restrict__ slot,
                                 float* __restrict__ out) {
  int i = blockIdx.x * blockDim.x + threadIdx.x;
  if (i < Bn * Kn * Dsn) out[i] = slot[i];
}

__global__ void vis_kernel(const float* __restrict__ attn_buf,
                           float* __restrict__ out_vis) {
  __shared__ float red[8];
  const int bk = blockIdx.x;
  float mn = 3.4e38f, mx = -3.4e38f;
  for (int n = threadIdx.x; n < Nn; n += 256) {
    float a = attn_buf[(size_t)bk * Nn + n];
    mn = fminf(mn, a); mx = fmaxf(mx, a);
  }
  mn = block_min256(mn, red);
  mx = block_max256(mx, red);
  const float inv = 1.0f / (mx - mn + 1e-5f);
  for (int n = threadIdx.x; n < Nn; n += 256) {
    float a = attn_buf[(size_t)bk * Nn + n];
    out_vis[(size_t)bk * Nn + n] = (a - mn) * inv;
  }
}

// ---------------- host ----------------
static inline size_t align256(size_t x) { return (x + 255) & ~(size_t)255; }

extern "C" void kernel_launch(void* const* d_in, const int* in_sizes, int n_in,
                              void* d_out, int out_size, void* d_ws, size_t ws_size,
                              hipStream_t stream) {
  const float* feat       = (const float*)d_in[0];
  const float* mask       = (const float*)d_in[1];
  const float* noise      = (const float*)d_in[2];
  const float* slots_mu   = (const float*)d_in[3];
  const float* slots_ls   = (const float*)d_in[4];
  const float* norm_g     = (const float*)d_in[5];
  const float* norm_b     = (const float*)d_in[6];
  const float* grid_W     = (const float*)d_in[7];
  const float* grid_b     = (const float*)d_in[8];
  const float* k_W        = (const float*)d_in[9];
  const float* v_W        = (const float*)d_in[10];
  const float* mlp_g      = (const float*)d_in[11];
  const float* mlp_b      = (const float*)d_in[12];
  const float* mlp_W      = (const float*)d_in[13];
  const float* mlp_bias   = (const float*)d_in[14];
  const float* q_g        = (const float*)d_in[15];
  const float* q_b        = (const float*)d_in[16];
  const float* q_W        = (const float*)d_in[17];
  const float* gru_Wih    = (const float*)d_in[18];
  const float* gru_Whh    = (const float*)d_in[19];
  const float* gru_bih    = (const float*)d_in[20];
  const float* gru_bhh    = (const float*)d_in[21];
  const float* res_g      = (const float*)d_in[22];
  const float* res_b      = (const float*)d_in[23];
  const float* res_W      = (const float*)d_in[24];
  const float* res_bias   = (const float*)d_in[25];

  float* out = (float*)d_out;
  float* out_slot = out;                       // (B,K,Ds) = 5120
  float* out_fg   = out + Bn * Kn * Dsn;       // (B,K,2)  = 40
  float* out_vis  = out_fg + Bn * Kn * 2;      // (B,K,N)  = 81920

  char* ws = (char*)d_ws;
  size_t off = 0;
  half_t* featn_h = (half_t*)(ws + off); off = align256(off + (size_t)BN * Cn * 2);
  half_t* kW_h    = (half_t*)(ws + off); off = align256(off + (size_t)Cn * Cn * 2);
  half_t* vW_h    = (half_t*)(ws + off); off = align256(off + (size_t)Cn * Cn * 2);
  half_t* mlpW_h  = (half_t*)(ws + off); off = align256(off + (size_t)Dsn * Cn * 2);
  half_t* kbase_h = (half_t*)(ws + off); off = align256(off + (size_t)BN * Cn * 2);
  half_t* vbase_h = (half_t*)(ws + off); off = align256(off + (size_t)BN * Cn * 2);
  half_t* kk_h    = (half_t*)(ws + off); off = align256(off + (size_t)BKN * Dsn * 2);
  half_t* vv_h    = (half_t*)(ws + off); off = align256(off + (size_t)BKN * Dsn * 2);
  float*  attn_f  = (float*)(ws + off);  off = align256(off + (size_t)BKN * 4);
  float*  slot_f  = (float*)(ws + off);  off = align256(off + (size_t)Bn * Kn * Dsn * 4);
  float*  upd_f   = (float*)(ws + off);  off = align256(off + (size_t)Bn * Kn * Dsn * 4);
  float*  q_f     = (float*)(ws + off);  off = align256(off + (size_t)Bn * Kn * Dsn * 4);
  float*  fgpos_f = (float*)(ws + off);  off = align256(off + (size_t)Kn * 2 * 4);
  (void)ws_size; (void)n_in; (void)in_sizes; (void)out_size;

  // 1. weight conversion to f16
  cvt_f32_to_f16<<<(Cn * Cn + 255) / 256, 256, 0, stream>>>(k_W, kW_h, Cn * Cn);
  cvt_f32_to_f16<<<(Cn * Cn + 255) / 256, 256, 0, stream>>>(v_W, vW_h, Cn * Cn);
  cvt_f32_to_f16<<<(Dsn * Cn + 255) / 256, 256, 0, stream>>>(mlp_W, mlpW_h, Dsn * Cn);

  // 2. LN(feat) -> f16
  ln_feat_kernel<<<BN, 256, 0, stream>>>(feat, norm_g, norm_b, featn_h);

  // 3. WMMA GEMMs: kbase / vbase = feat_n @ W.T
  dim3 ggrid(BN / 32, Cn / 64);
  gemm_wt_f16<<<ggrid, 256, 0, stream>>>(featn_h, kW_h, kbase_h, Cn, Cn);
  gemm_wt_f16<<<ggrid, 256, 0, stream>>>(featn_h, vW_h, vbase_h, Cn, Cn);

  // 4. fg_pos (also fills d_out fg section)
  fgpos_kernel<<<Kn, 256, 0, stream>>>(mask, fgpos_f, out_fg);

  // 5. slot init
  slot_init_kernel<<<(Bn * Kn * Dsn + 255) / 256, 256, 0, stream>>>(
      slots_mu, slots_ls, noise, slot_f);

  // 6. fused pos-embed + LN + WMMA MLP -> kk, vv
  kkvv_kernel<<<BKN / 16, 256, 0, stream>>>(kbase_h, mlpW_h, grid_W, grid_b,
                                            mlp_g, mlp_b, mlp_bias, fgpos_f, kk_h);
  kkvv_kernel<<<BKN / 16, 256, 0, stream>>>(vbase_h, mlpW_h, grid_W, grid_b,
                                            mlp_g, mlp_b, mlp_bias, fgpos_f, vv_h);

  // 7. slot attention iterations
  for (int it = 0; it < 4; ++it) {
    q_kernel<<<Bn * Kn, 256, 0, stream>>>(slot_f, q_g, q_b, q_W, q_f);
    attn_kernel<<<Bn * Kn, 256, 0, stream>>>(q_f, kk_h, vv_h, mask, attn_f, upd_f);
    gru_kernel<<<Bn * Kn, 256, 0, stream>>>(upd_f, slot_f, gru_Wih, gru_Whh,
                                            gru_bih, gru_bhh, res_g, res_b,
                                            res_W, res_bias, slot_f);
  }

  // 8. outputs
  copy_slot_kernel<<<(Bn * Kn * Dsn + 255) / 256, 256, 0, stream>>>(slot_f, out_slot);
  vis_kernel<<<Bn * Kn, 256, 0, stream>>>(attn_f, out_vis);
}
